// AttentionLayer_49589692400163
// MI455X (gfx1250) — compile-verified, matrix operand order checked by reading, not run on hardware
//
#include <hip/hip_runtime.h>

// Problem constants (from reference)
#define BN 4096
#define SN 200
#define EN 128
#define HN 64
#define GN 32
#define SPAD 208   // 13 tiles of 16
#define NTILES 13

typedef __attribute__((ext_vector_type(16))) _Float16 v16h;
typedef __attribute__((ext_vector_type(8)))  _Float16 v8h;
typedef __attribute__((ext_vector_type(8)))  float    v8f;
typedef __attribute__((ext_vector_type(4)))  float    v4f;

// K-offset pattern within a 16-bit WMMA A/B fragment (ISA 7.12.2, 16-bit A 16x32):
// lane<16: halves 0..7 -> K 0..7, halves 8..15 -> K 16..23
// lane>=16: halves 0..7 -> K 8..15, halves 8..15 -> K 24..31
__device__ __forceinline__ int koff(int j, int lhalf) {
    return (j < 8) ? (lhalf * 8 + j) : (16 + lhalf * 8 + (j - 8));
}

__global__ __launch_bounds__(128) void attn_mlp_kernel(
    const float* __restrict__ q,     // [B,E]
    const float* __restrict__ keys,  // [B,S,E]
    const int*   __restrict__ mask,  // [B,S]
    const float* __restrict__ W1,    // [2E,H]
    const float* __restrict__ b1,    // [H]
    const float* __restrict__ W2,    // [H,G]
    const float* __restrict__ b2,    // [G]
    const float* __restrict__ W3,    // [G,1]
    const float* __restrict__ b3,    // [1]
    float* __restrict__ out_att,     // [B,E]
    float* __restrict__ out_w)       // [B,S]
{
    // ---- LDS ----
    __shared__ __align__(32) _Float16 sW1[4 * 4 * 32 * 16]; // [ks][nt][lane][j] pre-swizzled W1_bot fragments (16 KB)
    __shared__ __align__(32) _Float16 sW2[2 * 2 * 32 * 16]; // [ks][nt][lane][j] W2 fragments (4 KB)
    __shared__ float sQh1[HN];                              // q*W1_top + b1
    __shared__ __align__(32) _Float16 sH[4][16 * 64];       // per-wave h1 (f16) / reused as h2 (f32)
    __shared__ float sScore[SPAD];
    __shared__ float sRed[128];

    const int b    = blockIdx.x;
    const int tid  = threadIdx.x;
    const int lane = tid & 31;
    const int wave = tid >> 5;
    const int lhf  = lane >> 4;   // lane half (0/1)
    const int ln   = lane & 15;

    // ---- Phase 0: stage weights into LDS in WMMA fragment order + per-b q*W1_top ----
    for (int idx = tid; idx < 4 * 4 * 32 * 16; idx += 128) {
        int j = idx & 15, l = (idx >> 4) & 31, nt = (idx >> 9) & 3, ks = idx >> 11;
        int k = ks * 32 + koff(j, l >> 4);
        int n = nt * 16 + (l & 15);
        sW1[idx] = (_Float16)W1[(size_t)(EN + k) * HN + n];   // bottom half of W1 (keys part)
    }
    for (int idx = tid; idx < 2 * 2 * 32 * 16; idx += 128) {
        int j = idx & 15, l = (idx >> 4) & 31, nt = (idx >> 9) & 1, ks = (idx >> 10) & 1;
        int k = ks * 32 + koff(j, l >> 4);
        int n = nt * 16 + (l & 15);
        sW2[idx] = (_Float16)W2[(size_t)k * GN + n];
    }
    if (tid < HN) {
        float acc = b1[tid];
        const float* qp = q + (size_t)b * EN;
        #pragma unroll 4
        for (int e = 0; e < EN; ++e) acc = fmaf(qp[e], W1[(size_t)e * HN + tid], acc);
        sQh1[tid] = acc;
    }
    __syncthreads();

    // ---- Phase 1: per-wave S-tiles through the MLP ----
    const float* keyb = keys + (size_t)b * SN * EN;
    float qv[4];
    #pragma unroll
    for (int nt = 0; nt < 4; ++nt) qv[nt] = sQh1[nt * 16 + ln];
    float b2v[2];
    b2v[0] = b2[ln]; b2v[1] = b2[16 + ln];
    const float b3v = b3[0];
    float w3r[GN];
    #pragma unroll
    for (int n = 0; n < GN; ++n) w3r[n] = W3[n];

    const v16h Z16 = {};
    for (int t = wave; t < NTILES; t += 4) {
        // prefetch next tile's rows
        if (t + 4 < NTILES) {
            int ps = (t + 4) * 16 + ln;
            if (ps < SN) __builtin_prefetch(keyb + (size_t)ps * EN, 0, 3);
        }
        v8f acc[4] = {};
        const int srow = t * 16 + ln;
        #pragma unroll
        for (int ks = 0; ks < 4; ++ks) {
            // A fragment: 16 rows of keys, K-slice ks*32..ks*32+31
            v16h A = Z16;
            if (srow < SN) {
                const float* p = keyb + (size_t)srow * EN + ks * 32 + lhf * 8;
                v4f a0 = *(const v4f*)(p);
                v4f a1 = *(const v4f*)(p + 4);
                v4f a2 = *(const v4f*)(p + 16);
                v4f a3 = *(const v4f*)(p + 20);
                #pragma unroll
                for (int j = 0; j < 4; ++j) {
                    A[j]      = (_Float16)a0[j];
                    A[4 + j]  = (_Float16)a1[j];
                    A[8 + j]  = (_Float16)a2[j];
                    A[12 + j] = (_Float16)a3[j];
                }
            }
            #pragma unroll
            for (int nt = 0; nt < 4; ++nt) {
                v16h Bf = *(const v16h*)&sW1[((ks * 4 + nt) * 32 + lane) * 16];
                acc[nt] = __builtin_amdgcn_wmma_f32_16x16x32_f16(
                    false, A, false, Bf, (short)0, acc[nt], false, false);
            }
        }
        // h1 = relu(k*W1_bot + (q*W1_top + b1)); store f16 to per-wave LDS buffer
        _Float16* hb = &sH[wave][0];
        #pragma unroll
        for (int nt = 0; nt < 4; ++nt) {
            int col = nt * 16 + ln;
            #pragma unroll
            for (int r = 0; r < 8; ++r) {
                float v = acc[nt][r] + qv[nt];
                v = v > 0.f ? v : 0.f;
                hb[(r + lhf * 8) * 64 + col] = (_Float16)v;
            }
        }
        asm volatile("s_wait_dscnt 0" ::: "memory");

        // Layer 2: A fragments (16x64 f16) from LDS, contiguous 16B runs
        v16h A2[2];
        #pragma unroll
        for (int ks = 0; ks < 2; ++ks) {
            const _Float16* pr = hb + ln * 64 + ks * 32 + lhf * 8;
            v8h lo = *(const v8h*)pr;
            v8h hi = *(const v8h*)(pr + 16);
            #pragma unroll
            for (int j = 0; j < 8; ++j) { A2[ks][j] = lo[j]; A2[ks][8 + j] = hi[j]; }
        }
        asm volatile("" ::: "memory");
        v8f acc2[2] = {};
        #pragma unroll
        for (int ks = 0; ks < 2; ++ks)
            #pragma unroll
            for (int nt = 0; nt < 2; ++nt) {
                v16h Bf = *(const v16h*)&sW2[((ks * 2 + nt) * 32 + lane) * 16];
                acc2[nt] = __builtin_amdgcn_wmma_f32_16x16x32_f16(
                    false, A2[ks], false, Bf, (short)0, acc2[nt], false, false);
            }
        // h2 = relu(. + b2); reuse per-wave buffer as f32 [16][32]
        float* h2b = (float*)hb;
        #pragma unroll
        for (int nt = 0; nt < 2; ++nt) {
            int col = nt * 16 + ln;
            #pragma unroll
            for (int r = 0; r < 8; ++r) {
                float v = acc2[nt][r] + b2v[nt];
                v = v > 0.f ? v : 0.f;
                h2b[(r + lhf * 8) * 32 + col] = v;
            }
        }
        asm volatile("s_wait_dscnt 0" ::: "memory");

        // Layer 3: per-row 32-wide dot + bias, mask, scatter to scores
        if (lane < 16) {
            float sc = b3v;
            #pragma unroll
            for (int n = 0; n < GN; ++n) sc = fmaf(h2b[lane * 32 + n], w3r[n], sc);
            int s = t * 16 + lane;
            if (s < SN) {
                if (mask[(size_t)b * SN + s] == 0) sc = -1e9f;
                sScore[s] = sc;
            } else {
                sScore[s] = -1e9f;   // padded rows get zero softmax weight
            }
        }
        asm volatile("s_wait_dscnt 0" ::: "memory");
    }
    __syncthreads();

    // ---- Phase 2: softmax over SPAD entries (128 threads) ----
    float m0 = sScore[tid];
    float m1 = (tid + 128 < SPAD) ? sScore[tid + 128] : -3.0e38f;
    sRed[tid] = fmaxf(m0, m1);
    __syncthreads();
    #pragma unroll
    for (int off = 64; off > 0; off >>= 1) {
        if (tid < off) sRed[tid] = fmaxf(sRed[tid], sRed[tid + off]);
        __syncthreads();
    }
    const float maxv = sRed[0];
    __syncthreads();

    float e0 = __expf(m0 - maxv);
    float e1 = (tid + 128 < SPAD) ? __expf(m1 - maxv) : 0.f;
    sScore[tid] = e0;
    if (tid + 128 < SPAD) sScore[tid + 128] = e1;
    sRed[tid] = e0 + e1;
    __syncthreads();
    #pragma unroll
    for (int off = 64; off > 0; off >>= 1) {
        if (tid < off) sRed[tid] += sRed[tid + off];
        __syncthreads();
    }
    const float inv = 1.0f / sRed[0];
    __syncthreads();

    // normalize, emit weights
    for (int s = tid; s < SN; s += 128) {
        float w = sScore[s] * inv;
        sScore[s] = w;
        out_w[(size_t)b * SN + s] = w;
    }
    __syncthreads();

    // ---- Phase 3: attended[e] = sum_s w[s] * keys[b,s,e] (f32, coalesced) ----
    if (tid < EN) {
        float acc = 0.f;
        #pragma unroll 4
        for (int s = 0; s < SN; ++s)
            acc = fmaf(sScore[s], keyb[(size_t)s * EN + tid], acc);
        out_att[(size_t)b * EN + tid] = acc;
    }
}

extern "C" void kernel_launch(void* const* d_in, const int* in_sizes, int n_in,
                              void* d_out, int out_size, void* d_ws, size_t ws_size,
                              hipStream_t stream) {
    const float* q    = (const float*)d_in[0];
    const float* keys = (const float*)d_in[1];
    const int*   mask = (const int*)  d_in[2];
    const float* W1   = (const float*)d_in[3];
    const float* b1   = (const float*)d_in[4];
    const float* W2   = (const float*)d_in[5];
    const float* b2   = (const float*)d_in[6];
    const float* W3   = (const float*)d_in[7];
    const float* b3   = (const float*)d_in[8];

    float* out_att = (float*)d_out;                       // [B,E] first in return order
    float* out_w   = (float*)d_out + (size_t)BN * EN;     // [B,S] second

    attn_mlp_kernel<<<dim3(BN), dim3(128), 0, stream>>>(
        q, keys, mask, W1, b1, W2, b2, W3, b3, out_att, out_w);
}
